// kernel_generated_0_62904091017463
// MI455X (gfx1250) — matrix operand order repaired and checked by reading, not compiled, not run on hardware
//
#include <hip/hip_runtime.h>
#include <hip/hip_bf16.h>

typedef __attribute__((ext_vector_type(16))) _Float16 v16h;
typedef __attribute__((ext_vector_type(8)))  float    v8f;

#define HH   56
#define WW   56
#define CIN  128
#define COUT 128

// One workgroup per (n, h): reads x[n, :, (h-1)%56, :] once, writes out[n, :, h, :] once.
// 8 waves; wave wv owns output-channel tile jt=wv (16 channels), loops 4 w-tiles of 16.
// Input row arrives via global_load_async_to_lds_b128; output row leaves via
// global_store_async_from_lds_b128; the 18-tap grouped contraction runs on
// v_wmma_f32_16x16x32_f16 (f16 in, f32 accumulate).
__global__ __launch_bounds__(256)
void fused_groupconv_wmma(const float* __restrict__ x,
                          const float* __restrict__ Wt,
                          float* __restrict__ out) {
    __shared__ float lds_x[CIN * WW];   // input row; reused as output staging [c][w]
    __shared__ float lds_a[64 * WW];    // A[cc][w] = x[cc] + x[cc+64]

    const int h  = blockIdx.x;
    const int n  = blockIdx.y;
    const int hp = (h + HH - 1) % HH;          // final roll along h
    const int t    = threadIdx.x;
    const int lane = t & 31;
    const int wv   = t >> 5;
    const int hi   = lane >> 4;                // lane half (WMMA layouts)

    const unsigned lds_x_base = (unsigned)(size_t)lds_x;

    // ---- Phase 1: async DMA x[n, :, hp, :] (128 rows x 224B) into LDS ----
    {
        const unsigned long long xbase = (unsigned long long)(size_t)x;
        #pragma unroll
        for (int it = 0; it < 7; ++it) {
            int flat  = it * 256 + t;          // 1792 x 16B chunks
            int c     = flat / 14;
            int chunk = flat - c * 14;
            unsigned long long gaddr = xbase +
                ((unsigned long long)((n * CIN + c) * HH + hp) * WW + (unsigned)(chunk * 4)) * 4ull;
            unsigned laddr = lds_x_base + (unsigned)(c * WW * 4 + chunk * 16);
            asm volatile("global_load_async_to_lds_b128 %0, %1, off"
                         :: "v"(laddr), "v"(gaddr) : "memory");
        }
        asm volatile("s_wait_asynccnt 0" ::: "memory");
    }
    __syncthreads();

    // ---- Phase 2: factor-2 channel-sum pool into lds_a ----
    #pragma unroll
    for (int it = 0; it < 14; ++it) {
        int idx = it * 256 + t;                // 0..3583
        lds_a[idx] = lds_x[idx] + lds_x[idx + 64 * WW];
    }
    __syncthreads();

    // ---- Phase 3: pack B fragments for this wave's channel tile ----
    // K-index q = (js_rel 0..5, m 0..1, l 0..2) -> 36 values, padded into 2x K=32.
    const int jt = wv;                         // output channel tile: c = jt*16 + nn
    const int nn = lane & 15;
    const int jj = nn >> 2, ko = nn & 3;       // c = 4*(4*jt+jj) + ko
    v16h bfrag[2];
    #pragma unroll
    for (int f = 0; f < 2; ++f) {
        #pragma unroll
        for (int e = 0; e < 16; ++e) {
            int q = f * 32 + e + hi * 16;      // B layout: K = e + 16*hi
            float val = 0.f;
            if (q < 36) {
                int js_rel = q / 6;
                int rem    = q - js_rel * 6;
                int m      = rem / 3;
                int l      = rem - m * 3;
                int i      = js_rel - jj;      // group-window tap
                if (i >= 0 && i < 3) {
                    int j = jt * 4 + jj;
                    val = Wt[(((j * 3 + i) * 2 + m) * 3 + l) * 4 + ko];
                }
            }
            bfrag[f][e] = (_Float16)val;
        }
    }

    // ---- Phase 4: 4 w-tiles: pack A, 2x WMMA (f32 acc), stage D to LDS ----
    #pragma unroll
    for (int mt = 0; mt < 4; ++mt) {
        const int mm = lane & 15;
        const int w  = mt * 16 + mm;           // output w position (M index)
        v16h afrag[2];
        #pragma unroll
        for (int f = 0; f < 2; ++f) {
            #pragma unroll
            for (int e = 0; e < 16; ++e) {
                int K = e + hi * 16;                    // A layout: K = e + 16*hi
                int q = f * 32 + K;
                float val = 0.f;
                if (q < 36 && w < WW) {
                    int js_rel = q / 6;
                    int rem    = q - js_rel * 6;
                    int m      = rem / 3;
                    int l      = rem - m * 3;
                    int js     = jt * 4 - 1 + js_rel;   // source group (zero-padded)
                    int wq     = w + l - 1;             // unfold tap (zero-padded)
                    if (js >= 0 && js < 32 && wq >= 0 && wq < WW) {
                        int wsrc = (wq + WW - 1) % WW;  // circular roll along w
                        val = lds_a[(js * 2 + m) * WW + wsrc];
                    }
                }
                afrag[f][e] = (_Float16)val;
            }
        }
        v8f acc = {};
        acc = __builtin_amdgcn_wmma_f32_16x16x32_f16(false, afrag[0], false, bfrag[0],
                                                     (short)0, acc, false, false);
        acc = __builtin_amdgcn_wmma_f32_16x16x32_f16(false, afrag[1], false, bfrag[1],
                                                     (short)0, acc, false, false);
        // stage D -> lds_x (reused as out[c][w]); C/D layout: M = r + 8*hi
        int c = jt * 16 + nn;
        #pragma unroll
        for (int r = 0; r < 8; ++r) {
            int wo = mt * 16 + r + hi * 8;
            if (wo < WW) lds_x[c * WW + wo] = acc[r];
        }
    }
    __syncthreads();

    // ---- Phase 5: async DMA output row LDS -> global (coalesced 16B chunks) ----
    {
        const unsigned long long obase = (unsigned long long)(size_t)out;
        #pragma unroll
        for (int it = 0; it < 7; ++it) {
            int flat  = it * 256 + t;
            int c     = flat / 14;
            int chunk = flat - c * 14;
            unsigned long long gaddr = obase +
                ((unsigned long long)((n * COUT + c) * HH + h) * WW + (unsigned)(chunk * 4)) * 4ull;
            unsigned laddr = lds_x_base + (unsigned)(c * WW * 4 + chunk * 16);
            asm volatile("global_store_async_from_lds_b128 %0, %1, off"
                         :: "v"(gaddr), "v"(laddr) : "memory");
        }
        asm volatile("s_wait_asynccnt 0" ::: "memory");
    }
}

extern "C" void kernel_launch(void* const* d_in, const int* in_sizes, int n_in,
                              void* d_out, int out_size, void* d_ws, size_t ws_size,
                              hipStream_t stream) {
    const float* x  = (const float*)d_in[0];   // (128,128,56,56) f32
    const float* Wt = (const float*)d_in[1];   // (32,3,2,3,4)    f32
    float* out = (float*)d_out;                // (128,128,56,56) f32
    dim3 grid(HH, 128);                        // (h, n)
    fused_groupconv_wmma<<<grid, 256, 0, stream>>>(x, Wt, out);
}
